// DeepSeekBlock_22239340658922
// MI455X (gfx1250) — matrix operand order repaired and checked by reading, not compiled, not run on hardware
//
#include <hip/hip_runtime.h>
#include <hip/hip_bf16.h>
#include <math.h>

// ---------------- problem constants ----------------
#define S_  2048
#define H_  1024
#define NH_ 16
#define KVH_ 4
#define HD_ 64
#define E_  8
#define I_  4096

typedef __attribute__((ext_vector_type(16))) _Float16 v16h;
typedef __attribute__((ext_vector_type(8)))  _Float16 v8h;
typedef __attribute__((ext_vector_type(8)))  float    v8f;

union V16 { v16h v; v8h h[2]; };

static __device__ __forceinline__ v8f wmma_f16(v16h a, v16h b, v8f c) {
    return __builtin_amdgcn_wmma_f32_16x16x32_f16(false, a, false, b, (short)0, c, false, false);
}

// ---------------- RMSNorm: f32 in -> f16 out ----------------
__global__ __launch_bounds__(256) void k_rmsnorm(const float* __restrict__ x,
                                                 const float* __restrict__ w,
                                                 _Float16* __restrict__ out) {
    int row = blockIdx.x, tid = threadIdx.x;
    const float* xr = x + (size_t)row * H_;
    float4 xv = *(const float4*)(xr + tid * 4);
    float ss = xv.x * xv.x + xv.y * xv.y + xv.z * xv.z + xv.w * xv.w;
    __shared__ float red[256];
    red[tid] = ss;
    for (int s = 128; s > 0; s >>= 1) { __syncthreads(); if (tid < s) red[tid] += red[tid + s]; }
    __syncthreads();
    float rs = rsqrtf(red[0] / (float)H_ + 1e-6f);
    _Float16* o = out + (size_t)row * H_ + tid * 4;
    const float* wp = w + tid * 4;
    o[0] = (_Float16)(wp[0] * xv.x * rs);
    o[1] = (_Float16)(wp[1] * xv.y * rs);
    o[2] = (_Float16)(wp[2] * xv.z * rs);
    o[3] = (_Float16)(wp[3] * xv.w * rs);
}

// ---------------- RoPE in-place on f16 [S, nh*64] ----------------
__global__ void k_rope(_Float16* __restrict__ x, int nh) {
    size_t idx = (size_t)blockIdx.x * blockDim.x + threadIdx.x;
    size_t total = (size_t)S_ * nh * 32;
    if (idx >= total) return;
    int d = (int)(idx & 31);
    size_t tmp = idx >> 5;
    int head = (int)(tmp % nh);
    int s = (int)(tmp / nh);
    size_t base = (size_t)s * nh * HD_ + (size_t)head * HD_ + d;
    float x1 = (float)x[base], x2 = (float)x[base + 32];
    float freq = (float)s * powf(10000.0f, -(float)d / 32.0f);
    float c = cosf(freq), sn = sinf(freq);
    x[base]      = (_Float16)(x1 * c - x2 * sn);
    x[base + 32] = (_Float16)(x2 * c + x1 * sn);
}

// ---------------- Tiled WMMA GEMM ----------------
// C[M,N] = A[M,K](f16, row-major, lda) @ B[K,N](f32, row-major, ldb=N)
// Block tile 64x128, K-step 32. 8 waves (2x4), each wave 32x32 (2x2 frags).
// A tile staged LDS via GLOBAL_LOAD_ASYNC_TO_LDS_B128 (ASYNCcnt path) when no
// gather; B tile f32->f16 converted + transposed [n][k] through VGPRs.
// epi: 0 store f16, 1 silu->f16, 2 acc+res -> out32 & out32b (f32),
//      3 out32 += acc, 4 scatter: out32[scatter[row]*N+col] += escale[row]*acc
// gather: optional A-row indirection (MoE dispatch; idx<0 -> zero row)
// live_cnt: optional row count for early block exit (MoE)
__global__ __launch_bounds__(256) void k_gemm_f16_wmma(
    const _Float16* __restrict__ A, const float* __restrict__ B,
    int M, int N, int K, int lda, int epi,
    _Float16* __restrict__ out16, float* __restrict__ out32,
    float* __restrict__ out32b, const float* __restrict__ res,
    const int* __restrict__ gather, const int* __restrict__ scatter,
    const float* __restrict__ escale, const int* __restrict__ live_cnt) {
    int bm = blockIdx.y, bn = blockIdx.x;
    if (live_cnt) {
        if (bm * 64 >= *live_cnt) return;   // uniform early-exit, before barriers
    }
    __shared__ __align__(32) _Float16 sA[64 * 32];    // [m][k]
    __shared__ __align__(32) _Float16 sB[128 * 32];   // B^T tile: [n][k]
    int tid = threadIdx.x, lane = tid & 31, wave = tid >> 5;
    int wm = wave >> 2, wn = wave & 3, half = lane >> 4, ln = lane & 15;

    v8f acc[2][2];
    #pragma unroll
    for (int i = 0; i < 2; i++)
        #pragma unroll
        for (int j = 0; j < 2; j++)
            #pragma unroll
            for (int r = 0; r < 8; r++) acc[i][j][r] = 0.0f;

    for (int k0 = 0; k0 < K; k0 += 32) {
        { // stage A (16B per thread)
            int row = tid >> 2, c0 = (tid & 3) * 8;
            int grow = bm * 64 + row;
            if (gather) {
                int src = gather[grow];
                v8h zv;
                #pragma unroll
                for (int x = 0; x < 8; x++) zv[x] = (_Float16)0.0f;
                if (src >= 0) zv = *(const v8h*)(A + (size_t)src * lda + k0 + c0);
                *(v8h*)&sA[row * 32 + c0] = zv;
            } else {
                const _Float16* ap = A + (size_t)grow * lda + k0 + c0;
                unsigned ldst = (unsigned)(unsigned long long)&sA[row * 32 + c0];
                asm volatile("global_load_async_to_lds_b128 %0, %1, off"
                             :: "v"(ldst), "v"(ap) : "memory");
            }
        }
        { // stage B transposed + f32->f16 convert
            int kk = tid >> 3, n0 = (tid & 7) * 16;
            const float* bp = B + (size_t)(k0 + kk) * N + bn * 128 + n0;
            #pragma unroll
            for (int x = 0; x < 16; x++) sB[(n0 + x) * 32 + kk] = (_Float16)bp[x];
            if (k0 + 32 < K) __builtin_prefetch(bp + (size_t)32 * N, 0, 1);
        }
        if (!gather) asm volatile("s_wait_asynccnt 0x0" ::: "memory");
        __syncthreads();

        V16 a[2];
        #pragma unroll
        for (int i = 0; i < 2; i++) {
            int r = (wm * 32 + i * 16 + ln) * 32;
            a[i].h[0] = *(v8h*)&sA[r + half * 16];
            a[i].h[1] = *(v8h*)&sA[r + half * 16 + 8];
        }
        v16h b[2];
        #pragma unroll
        for (int j = 0; j < 2; j++)
            b[j] = *(v16h*)&sB[(wn * 32 + j * 16 + ln) * 32 + half * 16];

        #pragma unroll
        for (int i = 0; i < 2; i++)
            #pragma unroll
            for (int j = 0; j < 2; j++)
                acc[i][j] = wmma_f16(a[i].v, b[j], acc[i][j]);
        __syncthreads();
    }

    // epilogue (C frag: VGPR r -> M = r + half*8; N = ln)
    #pragma unroll
    for (int i = 0; i < 2; i++) {
        #pragma unroll
        for (int j = 0; j < 2; j++) {
            int c0 = bn * 128 + wn * 32 + j * 16 + ln;
            #pragma unroll
            for (int r = 0; r < 8; r++) {
                int row = bm * 64 + wm * 32 + i * 16 + r + half * 8;
                float val = acc[i][j][r];
                size_t idx = (size_t)row * N + c0;
                if (epi == 0) {
                    out16[idx] = (_Float16)val;
                } else if (epi == 1) {
                    out16[idx] = (_Float16)(val / (1.0f + expf(-val)));
                } else if (epi == 2) {
                    float v2 = val + res[idx];
                    out32[idx] = v2;
                    out32b[idx] = v2;
                } else if (epi == 3) {
                    out32[idx] += val;
                } else { // 4: MoE scatter with per-slot weight
                    int tok = scatter[row];
                    if (tok >= 0)
                        out32[(size_t)tok * N + c0] += escale[row] * val;
                }
            }
        }
    }
}

// ---------------- Flash-style GQA attention ----------------
// grid (NH, S/64), block 128 (4 waves x 16 q-rows). Q pre-scaled by 1/8.
__global__ __launch_bounds__(128) void k_attn_wmma(
    const _Float16* __restrict__ q, const _Float16* __restrict__ kbuf,
    const _Float16* __restrict__ vbuf, _Float16* __restrict__ obuf) {
    __shared__ __align__(32) _Float16 sVt[64 * 32];       // [hd][key]
    __shared__ __align__(32) _Float16 sP[4][16 * 32];     // per-wave P
    int tid = threadIdx.x, lane = tid & 31, wave = tid >> 5;
    int half = lane >> 4, ln = lane & 15;
    int h = blockIdx.x, qb = blockIdx.y * 64;
    int kvcol = (h >> 2) * HD_;          // GQA: kv head = h/4
    int qcol = h * HD_;
    int qrow0 = qb + wave * 16;
    int kvld = KVH_ * HD_;               // 256

    V16 qA[2];
    #pragma unroll
    for (int c = 0; c < 2; c++) {
        const _Float16* qp = q + (size_t)(qrow0 + ln) * (NH_ * HD_) + qcol + c * 32;
        v8h lo = *(const v8h*)(qp + half * 16);
        v8h hi = *(const v8h*)(qp + half * 16 + 8);
        #pragma unroll
        for (int x = 0; x < 8; x++) { lo[x] = lo[x] * (_Float16)0.125f; hi[x] = hi[x] * (_Float16)0.125f; }
        qA[c].h[0] = lo; qA[c].h[1] = hi;
    }

    float m[8], l[8];
    v8f oacc[4];
    #pragma unroll
    for (int r = 0; r < 8; r++) { m[r] = -1.0e30f; l[r] = 0.0f; }
    #pragma unroll
    for (int n = 0; n < 4; n++)
        #pragma unroll
        for (int r = 0; r < 8; r++) oacc[n][r] = 0.0f;

    int qmax = qrow0 + 15;
    for (int kc = 0; kc < qb + 64; kc += 32) {
        { // cooperative V-transpose stage: sVt[hd][key]
            int key = tid >> 2, h0 = (tid & 3) * 16;
            const _Float16* vp = vbuf + (size_t)(kc + key) * kvld + kvcol + h0;
            v16h vv = *(const v16h*)vp;
            #pragma unroll
            for (int x = 0; x < 16; x++) sVt[(h0 + x) * 32 + key] = vv[x];
        }
        __syncthreads();

        if (kc <= qmax) {
            v8f s[2];
            #pragma unroll
            for (int jn = 0; jn < 2; jn++) {
                #pragma unroll
                for (int r = 0; r < 8; r++) s[jn][r] = 0.0f;
                const _Float16* kr = kbuf + (size_t)(kc + jn * 16 + ln) * kvld + kvcol;
                v16h kb0 = *(const v16h*)(kr + half * 16);
                v16h kb1 = *(const v16h*)(kr + 32 + half * 16);
                s[jn] = wmma_f16(qA[0].v, kb0, s[jn]);
                s[jn] = wmma_f16(qA[1].v, kb1, s[jn]);
            }
            #pragma unroll
            for (int r = 0; r < 8; r++) {
                int row = qrow0 + r + half * 8;
                #pragma unroll
                for (int jn = 0; jn < 2; jn++) {
                    int col = kc + jn * 16 + ln;
                    if (col > row) s[jn][r] += -1.0e9f;
                }
                float mx = fmaxf(s[0][r], s[1][r]);
                for (int off = 8; off >= 1; off >>= 1) mx = fmaxf(mx, __shfl_xor(mx, off, 16));
                float mn = fmaxf(m[r], mx);
                float corr = expf(m[r] - mn);
                float p0 = expf(s[0][r] - mn), p1 = expf(s[1][r] - mn);
                float rs = p0 + p1;
                for (int off = 8; off >= 1; off >>= 1) rs += __shfl_xor(rs, off, 16);
                l[r] = l[r] * corr + rs;
                m[r] = mn;
                #pragma unroll
                for (int n = 0; n < 4; n++) oacc[n][r] *= corr;
                sP[wave][(r + half * 8) * 32 + ln]       = (_Float16)p0;
                sP[wave][(r + half * 8) * 32 + 16 + ln]  = (_Float16)p1;
            }
            V16 pA;
            pA.h[0] = *(v8h*)&sP[wave][ln * 32 + half * 16];
            pA.h[1] = *(v8h*)&sP[wave][ln * 32 + half * 16 + 8];
            #pragma unroll
            for (int n = 0; n < 4; n++) {
                v16h vb = *(v16h*)&sVt[(n * 16 + ln) * 32 + half * 16];
                oacc[n] = wmma_f16(pA.v, vb, oacc[n]);
            }
        }
        __syncthreads();
    }

    #pragma unroll
    for (int n = 0; n < 4; n++)
        #pragma unroll
        for (int r = 0; r < 8; r++) {
            int row = qrow0 + r + half * 8;
            int col = qcol + n * 16 + ln;
            obuf[(size_t)row * (NH_ * HD_) + col] = (_Float16)(oacc[n][r] / l[r]);
        }
}

// ---------------- MoE dispatch init ----------------
__global__ __launch_bounds__(256) void k_init_dispatch(int* __restrict__ cnt,
                                                       int* __restrict__ elist) {
    int idx = blockIdx.x * 256 + threadIdx.x;
    if (idx < E_) cnt[idx] = 0;
    if (idx < E_ * S_) elist[idx] = -1;
}

// ---------------- Router: softmax + top-2, build per-expert gather lists ----
__global__ __launch_bounds__(256) void k_router(const _Float16* __restrict__ x,
                                                const float* __restrict__ rw,
                                                int* __restrict__ cnt,
                                                int* __restrict__ elist,
                                                float* __restrict__ escale) {
    int tid = threadIdx.x, lane = tid & 31, wave = tid >> 5;
    int t = blockIdx.x * 8 + wave;
    float acc[E_];
    #pragma unroll
    for (int e = 0; e < E_; e++) acc[e] = 0.0f;
    for (int i = 0; i < H_ / 32; i++) {
        int hh = i * 32 + lane;
        float xv = (float)x[(size_t)t * H_ + hh];
        const float* wr = rw + (size_t)hh * E_;
        #pragma unroll
        for (int e = 0; e < E_; e++) acc[e] += xv * wr[e];
    }
    #pragma unroll
    for (int e = 0; e < E_; e++)
        for (int off = 16; off >= 1; off >>= 1) acc[e] += __shfl_xor(acc[e], off, 32);
    if (lane == 0) {
        float mx = acc[0];
        #pragma unroll
        for (int e = 1; e < E_; e++) mx = fmaxf(mx, acc[e]);
        float p[E_], ssum = 0.0f;
        #pragma unroll
        for (int e = 0; e < E_; e++) { p[e] = expf(acc[e] - mx); ssum += p[e]; }
        #pragma unroll
        for (int e = 0; e < E_; e++) p[e] /= ssum;
        int i1 = 0;
        #pragma unroll
        for (int e = 1; e < E_; e++) if (p[e] > p[i1]) i1 = e;
        int i2 = (i1 == 0) ? 1 : 0;
        #pragma unroll
        for (int e = 0; e < E_; e++) if (e != i1 && p[e] > p[i2]) i2 = e;
        float w1 = p[i1], w2 = p[i2], sw = w1 + w2;
        int p1 = atomicAdd(&cnt[i1], 1);
        elist[i1 * S_ + p1] = t;
        escale[i1 * S_ + p1] = w1 / sw;
        int p2 = atomicAdd(&cnt[i2], 1);
        elist[i2 * S_ + p2] = t;
        escale[i2 * S_ + p2] = w2 / sw;
    }
}

// ---------------- launch ----------------
extern "C" void kernel_launch(void* const* d_in, const int* in_sizes, int n_in,
                              void* d_out, int out_size, void* d_ws, size_t ws_size,
                              hipStream_t stream) {
    (void)in_sizes; (void)n_in; (void)out_size; (void)ws_size;
    const float* hidden = (const float*)d_in[0];
    const float* ln1w   = (const float*)d_in[1];
    const float* wq     = (const float*)d_in[2];
    const float* wk     = (const float*)d_in[3];
    const float* wv     = (const float*)d_in[4];
    const float* wo     = (const float*)d_in[5];
    const float* ln2w   = (const float*)d_in[6];
    const float* sw1    = (const float*)d_in[7];
    const float* sw2    = (const float*)d_in[8];
    const float* ew1    = (const float*)d_in[9];
    const float* ew2    = (const float*)d_in[10];
    const float* rw     = (const float*)d_in[11];
    float* out = (float*)d_out;

    char* w = (char*)d_ws;
    size_t off = 0;
    auto alloc = [&](size_t bytes) -> void* {
        void* p = w + off;
        off = (off + bytes + 255) & ~(size_t)255;
        return p;
    };
    _Float16* x1   = (_Float16*)alloc((size_t)S_ * H_ * 2);
    _Float16* qb   = (_Float16*)alloc((size_t)S_ * NH_ * HD_ * 2);
    _Float16* kb   = (_Float16*)alloc((size_t)S_ * KVH_ * HD_ * 2);
    _Float16* vb   = (_Float16*)alloc((size_t)S_ * KVH_ * HD_ * 2);
    _Float16* ao   = (_Float16*)alloc((size_t)S_ * NH_ * HD_ * 2);
    float*    hid2 = (float*)   alloc((size_t)S_ * H_ * 4);
    _Float16* x2   = (_Float16*)alloc((size_t)S_ * H_ * 2);
    _Float16* gb   = (_Float16*)alloc((size_t)S_ * I_ * 2);
    int*      cnt  = (int*)     alloc((size_t)E_ * 4);
    int*      elst = (int*)     alloc((size_t)E_ * S_ * 4);
    float*    escl = (float*)   alloc((size_t)E_ * S_ * 4);

    auto gemm = [&](const _Float16* A, const float* Bm, int M, int N, int K, int lda,
                    int epi, _Float16* o16, float* o32, float* o32b, const float* res,
                    const int* gath, const int* scat, const float* esc, const int* live) {
        dim3 g(N / 128, M / 64);
        k_gemm_f16_wmma<<<g, 256, 0, stream>>>(A, Bm, M, N, K, lda, epi,
                                               o16, o32, o32b, res, gath, scat, esc, live);
    };

    // 1) ln1 -> x1 (f16)
    k_rmsnorm<<<S_, 256, 0, stream>>>(hidden, ln1w, x1);
    // 2) QKV projections
    gemm(x1, wq, S_, NH_ * HD_, H_, H_, 0, qb, nullptr, nullptr, nullptr,
         nullptr, nullptr, nullptr, nullptr);
    gemm(x1, wk, S_, KVH_ * HD_, H_, H_, 0, kb, nullptr, nullptr, nullptr,
         nullptr, nullptr, nullptr, nullptr);
    gemm(x1, wv, S_, KVH_ * HD_, H_, H_, 0, vb, nullptr, nullptr, nullptr,
         nullptr, nullptr, nullptr, nullptr);
    // 3) RoPE on q, k
    {
        size_t tq = (size_t)S_ * NH_ * 32, tk = (size_t)S_ * KVH_ * 32;
        k_rope<<<(unsigned)((tq + 255) / 256), 256, 0, stream>>>(qb, NH_);
        k_rope<<<(unsigned)((tk + 255) / 256), 256, 0, stream>>>(kb, KVH_);
    }
    // 4) attention
    k_attn_wmma<<<dim3(NH_, S_ / 64), 128, 0, stream>>>(qb, kb, vb, ao);
    // 5) O-proj + residual -> hid2 and d_out
    gemm(ao, wo, S_, H_, NH_ * HD_, NH_ * HD_, 2, nullptr, hid2, out, hidden,
         nullptr, nullptr, nullptr, nullptr);
    // 6) ln2 -> x2
    k_rmsnorm<<<S_, 256, 0, stream>>>(hid2, ln2w, x2);
    // 7) shared MLP
    gemm(x2, sw1, S_, I_, H_, H_, 1, gb, nullptr, nullptr, nullptr,
         nullptr, nullptr, nullptr, nullptr);
    gemm(gb, sw2, S_, H_, I_, I_, 3, nullptr, out, nullptr, nullptr,
         nullptr, nullptr, nullptr, nullptr);
    // 8) router -> per-expert gather lists (top-2)
    k_init_dispatch<<<(E_ * S_ + 255) / 256, 256, 0, stream>>>(cnt, elst);
    k_router<<<S_ / 8, 256, 0, stream>>>(x2, rw, cnt, elst, escl);
    // 9) experts: gathered GEMM1 (silu) + scatter-weighted GEMM2, with
    //    per-expert early block exit (work ~ routed tokens, not S*E)
    for (int e = 0; e < E_; e++) {
        gemm(x2, ew1 + (size_t)e * H_ * I_, S_, I_, H_, H_, 1,
             gb, nullptr, nullptr, nullptr, elst + e * S_, nullptr, nullptr, cnt + e);
        gemm(gb, ew2 + (size_t)e * I_ * H_, S_, H_, I_, I_, 4,
             nullptr, out, nullptr, nullptr, nullptr, elst + e * S_, escl + e * S_, cnt + e);
    }
}
